// MultiHeadSelfAttention_11974368821386
// MI455X (gfx1250) — compile-verified, matrix operand order checked by reading, not run on hardware
//
#include <hip/hip_runtime.h>
#include <hip/hip_bf16.h>

// ---------------------------------------------------------------------------
// MHA forward on gfx1250 (CDNA5).  All matmuls run on v_wmma_f32_16x16x32_f16.
//   B=4, S=2048, D=2048, H=16, hd=128.
// Pipeline:
//   1) gemm_qkv : x @ {Wq,Wk,Wv} + b  -> f16 Q(scaled)/K/V in [B,H,S,hd]
//   2) flash_attn: causal online-softmax attention (QK^T and P*V on WMMA)
//   3) gemm_out : ctx(f16) @ Wo + bo  -> f32 output
// Staging paths use 128-bit global loads; fragments come from LDS as b128.
// ---------------------------------------------------------------------------

typedef __attribute__((ext_vector_type(16))) _Float16 v16h;
typedef __attribute__((ext_vector_type(8)))  _Float16 v8h;
typedef __attribute__((ext_vector_type(4)))  _Float16 v4h;
typedef __attribute__((ext_vector_type(8)))  float    v8f;
typedef __attribute__((ext_vector_type(4)))  float    v4f;

#define D_MODEL 2048
#define N_HEADS 16
#define HEAD_DIM 128
#define SEQ 2048
#define BATCH 4
#define M_ROWS (BATCH * SEQ)   // 8192

__device__ __forceinline__ v8f zero8() {
  v8f z = {0.f, 0.f, 0.f, 0.f, 0.f, 0.f, 0.f, 0.f};
  return z;
}

__device__ __forceinline__ v8f wmma_f16(v16h a, v16h b, v8f c) {
  // (neg_a, A, neg_b, B, c_mod, C, reuse_a, reuse_b)
  return __builtin_amdgcn_wmma_f32_16x16x32_f16(false, a, false, b, (short)0, c,
                                                false, false);
}

// Load a 16x32 f16 fragment (A-layout; identical math serves B with n<->m).
// ISA 7.12.2: lanes 0-15 : row = lane,   K = {0..7, 16..23}
//             lanes 16-31: row = lane-16,K = {8..15, 24..31}
// p points at element (row0, k0) of a row-major [rows x >=32] tile, stride ld.
__device__ __forceinline__ v16h load_frag16(const _Float16* __restrict__ p, int ld) {
  const int lane = threadIdx.x & 31;
  const int r    = lane & 15;
  const int kh   = (lane >> 4) << 3;   // 0 or 8
  const _Float16* q = p + r * ld + kh;
  v16h f;
#pragma unroll
  for (int i = 0; i < 4; ++i) {        // K = kh + 2i, kh + 2i + 1
    f[2 * i]     = q[2 * i];
    f[2 * i + 1] = q[2 * i + 1];
  }
#pragma unroll
  for (int i = 0; i < 4; ++i) {        // K = 16 + kh + 2i, +1
    f[8 + 2 * i]     = q[16 + 2 * i];
    f[8 + 2 * i + 1] = q[16 + 2 * i + 1];
  }
  return f;
}

// ---------------------------------------------------------------------------
// Kernel 1: QKV projections.  grid = (N/128, M/128, 3), block = 256 (8 waves).
// Block tile 128x128, K-step 32.  Wave grid 4(M) x 2(N): wave tile 32x64.
// ---------------------------------------------------------------------------
__global__ __launch_bounds__(256) void gemm_qkv(
    const float* __restrict__ x,
    const float* __restrict__ Wq, const float* __restrict__ Wk,
    const float* __restrict__ Wv,
    const float* __restrict__ bq, const float* __restrict__ bk,
    const float* __restrict__ bv,
    _Float16* __restrict__ Qh, _Float16* __restrict__ Kh,
    _Float16* __restrict__ Vh) {
  constexpr int BM = 128, BN = 128, BK = 32, LDK = BK + 8;  // 40 halves = 80 B
  __shared__ _Float16 sA[BM * LDK];   // [m][k]
  __shared__ _Float16 sB[BN * LDK];   // [n][k] (W transposed into LDS)

  const int z = blockIdx.z;
  const float* W    = (z == 0) ? Wq : (z == 1) ? Wk : Wv;
  const float* bias = (z == 0) ? bq : (z == 1) ? bk : bv;
  _Float16*    Oh   = (z == 0) ? Qh : (z == 1) ? Kh : Vh;
  const float  scl  = (z == 0) ? 0.08838834764831845f : 1.0f;  // 1/sqrt(128)

  const int m0 = blockIdx.y * BM;
  const int n0 = blockIdx.x * BN;
  const int tid = threadIdx.x;
  const int wave = tid >> 5, lane = tid & 31;
  const int wm = wave >> 1;            // 0..3 -> 32-row slab
  const int wn = wave & 1;             // 0..1 -> 64-col slab

  v8f acc[2][4];
#pragma unroll
  for (int i = 0; i < 2; ++i)
#pragma unroll
    for (int j = 0; j < 4; ++j) acc[i][j] = zero8();

  for (int kt = 0; kt < D_MODEL / BK; ++kt) {
    const int k0 = kt * BK;
    // Stage A (f32 -> f16): 128x32 elems as 1024 float4 (b128) loads.
#pragma unroll
    for (int i = 0; i < 4; ++i) {
      const int idx = tid + i * 256;         // 0..1023
      const int r = idx >> 3;                // 128 rows
      const int c = (idx & 7) * 4;           // 0,4,...,28
      const v4f a4 = *(const v4f*)&x[(size_t)(m0 + r) * D_MODEL + (k0 + c)];
      v4h h4;
      h4[0] = (_Float16)a4[0]; h4[1] = (_Float16)a4[1];
      h4[2] = (_Float16)a4[2]; h4[3] = (_Float16)a4[3];
      *(v4h*)&sA[r * LDK + c] = h4;          // 8-B aligned
    }
    // Stage B transposed (f32 -> f16): 32x128 elems as float4 loads.
#pragma unroll
    for (int i = 0; i < 4; ++i) {
      const int idx = tid + i * 256;
      const int k = idx >> 5;                // 32 k-rows
      const int n = (idx & 31) * 4;          // 0,4,...,124
      const v4f b4 = *(const v4f*)&W[(size_t)(k0 + k) * D_MODEL + (n0 + n)];
      sB[(n + 0) * LDK + k] = (_Float16)b4[0];
      sB[(n + 1) * LDK + k] = (_Float16)b4[1];
      sB[(n + 2) * LDK + k] = (_Float16)b4[2];
      sB[(n + 3) * LDK + k] = (_Float16)b4[3];
    }
    if (kt + 1 < D_MODEL / BK) {
      __builtin_prefetch(&W[(size_t)(k0 + BK) * D_MODEL + n0 + (tid & 127)], 0, 3);
      __builtin_prefetch(&x[(size_t)(m0 + (tid >> 1)) * D_MODEL + k0 + BK], 0, 3);
    }
    __syncthreads();

    v16h af[2], bf[4];
#pragma unroll
    for (int mi = 0; mi < 2; ++mi)
      af[mi] = load_frag16(&sA[(wm * 32 + mi * 16) * LDK], LDK);
#pragma unroll
    for (int ni = 0; ni < 4; ++ni)
      bf[ni] = load_frag16(&sB[(wn * 64 + ni * 16) * LDK], LDK);
#pragma unroll
    for (int mi = 0; mi < 2; ++mi)
#pragma unroll
      for (int ni = 0; ni < 4; ++ni)
        acc[mi][ni] = wmma_f16(af[mi], bf[ni], acc[mi][ni]);
    __syncthreads();
  }

  // Epilogue: bias, (Q scale), scatter to [B,H,S,hd] as f16.
#pragma unroll
  for (int mi = 0; mi < 2; ++mi)
#pragma unroll
    for (int ni = 0; ni < 4; ++ni) {
      const int fr = m0 + wm * 32 + mi * 16;
      const int fc = n0 + wn * 64 + ni * 16;
#pragma unroll
      for (int e = 0; e < 8; ++e) {
        const int row = fr + ((lane < 16) ? e : e + 8);  // b*S + s
        const int col = fc + (lane & 15);                // h*hd + d
        const float v = (acc[mi][ni][e] + bias[col]) * scl;
        const int b = row >> 11, s = row & 2047;
        const int h = col >> 7,  d = col & 127;
        Oh[(((size_t)(b * N_HEADS + h)) * SEQ + s) * HEAD_DIM + d] = (_Float16)v;
      }
    }
}

// ---------------------------------------------------------------------------
// Kernel 2: causal flash attention. grid = (S/64, B*H), block = 128 (4 waves).
// Each wave owns 16 query rows; block processes key tiles of 64 with online
// softmax.  Both QK^T and P*V are WMMA.
// ---------------------------------------------------------------------------
__global__ __launch_bounds__(128) void flash_attn(
    const _Float16* __restrict__ Qh, const _Float16* __restrict__ Kh,
    const _Float16* __restrict__ Vh, _Float16* __restrict__ Ctx) {
  constexpr int BQ = 64, BKV = 64;
  constexpr int LDK = HEAD_DIM + 8;   // 136 halves = 272 B (16-B aligned rows)
  constexpr int LDV = BKV + 8;        // sVt rows: [d][kpos]; sP rows: [q][kpos]
  constexpr int LDS_S = BKV + 4;
  __shared__ _Float16 sK[BKV * LDK];
  __shared__ _Float16 sVt[HEAD_DIM * LDV];
  __shared__ float    sS[BQ * LDS_S];
  __shared__ _Float16 sP[BQ * LDV];
  __shared__ float    sAlpha[BQ];
  __shared__ float    sL[BQ];

  const int qb = blockIdx.x;
  const int bh = blockIdx.y;
  const int b = bh >> 4, h = bh & 15;
  const size_t base = (size_t)bh * SEQ * HEAD_DIM;
  const int qrow0 = qb * BQ;
  const int tid = threadIdx.x;
  const int wave = tid >> 5, lane = tid & 31;

  // Q fragments for this wave's 16 rows, all 128 head dims (4 K-chunks).
  v16h qa[4];
#pragma unroll
  for (int c = 0; c < 4; ++c)
    qa[c] = load_frag16(Qh + base + (size_t)(qrow0 + wave * 16) * HEAD_DIM + c * 32,
                        HEAD_DIM);

  v8f o[8];
#pragma unroll
  for (int f = 0; f < 8; ++f) o[f] = zero8();
  float mrow = -3.0e38f, lrow = 0.0f;   // valid in threads 0..63 (rows)

  for (int jb = 0; jb <= qb; ++jb) {
    const int krow0 = jb * BKV;
    // Stage K ([kpos][d]) and V transposed ([d][kpos]) with b128 loads:
    // 64x128 halves = 1024 8-half vectors, 8 per thread.
#pragma unroll
    for (int i = 0; i < 8; ++i) {
      const int idx = tid + i * 128;         // 0..1023
      const int kp = idx >> 4;               // 64 key rows
      const int d  = (idx & 15) * 8;         // 0,8,...,120
      const size_t g = base + (size_t)(krow0 + kp) * HEAD_DIM + d;
      const v8h kv = *(const v8h*)&Kh[g];
      *(v8h*)&sK[kp * LDK + d] = kv;         // 16-B aligned
      const v8h vv = *(const v8h*)&Vh[g];
#pragma unroll
      for (int j = 0; j < 8; ++j) sVt[(d + j) * LDV + kp] = vv[j];
    }
    __syncthreads();

    // S = Q @ K^T  (16 x 64 per wave)
    v8f sc[4];
#pragma unroll
    for (int ni = 0; ni < 4; ++ni) sc[ni] = zero8();
#pragma unroll
    for (int ni = 0; ni < 4; ++ni)
#pragma unroll
      for (int c = 0; c < 4; ++c)
        sc[ni] = wmma_f16(qa[c], load_frag16(&sK[(ni * 16) * LDK + c * 32], LDK),
                          sc[ni]);

    // Spill scores to LDS for the row-wise softmax pass.
#pragma unroll
    for (int ni = 0; ni < 4; ++ni)
#pragma unroll
      for (int e = 0; e < 8; ++e) {
        const int r = wave * 16 + ((lane < 16) ? e : e + 8);
        const int c = ni * 16 + (lane & 15);
        sS[r * LDS_S + c] = sc[ni][e];
      }
    __syncthreads();

    // Online softmax: thread r owns query row r (r < 64).
    if (tid < BQ) {
      const int r = tid;
      int valid = qrow0 + r - krow0 + 1;          // causal
      valid = (valid > BKV) ? BKV : valid;
      const float* srow = &sS[r * LDS_S];
      float rmax = -3.0e38f;
      for (int c = 0; c < valid; ++c) rmax = fmaxf(rmax, srow[c]);
      const float mnew  = fmaxf(mrow, rmax);
      const float alpha = __expf(mrow - mnew);
      float ssum = 0.0f;
      for (int c = 0; c < BKV; ++c) {
        const float p = (c < valid) ? __expf(srow[c] - mnew) : 0.0f;
        sP[r * LDV + c] = (_Float16)p;
        ssum += p;
      }
      lrow = lrow * alpha + ssum;
      mrow = mnew;
      sAlpha[r] = alpha;
    }
    __syncthreads();

    // Rescale accumulated O by alpha (per fragment row), then O += P @ V.
    float ar[8];
#pragma unroll
    for (int e = 0; e < 8; ++e)
      ar[e] = sAlpha[wave * 16 + ((lane < 16) ? e : e + 8)];
#pragma unroll
    for (int f = 0; f < 8; ++f)
#pragma unroll
      for (int e = 0; e < 8; ++e) o[f][e] *= ar[e];

#pragma unroll
    for (int c = 0; c < 2; ++c) {                 // 64 key positions = 2 chunks
      const v16h pa = load_frag16(&sP[(wave * 16) * LDV + c * 32], LDV);
#pragma unroll
      for (int f = 0; f < 8; ++f)                 // 128 head dims = 8 N-tiles
        o[f] = wmma_f16(pa, load_frag16(&sVt[(f * 16) * LDV + c * 32], LDV), o[f]);
    }
    __syncthreads();
  }

  if (tid < BQ) sL[tid] = lrow;
  __syncthreads();

  float inv[8];
#pragma unroll
  for (int e = 0; e < 8; ++e)
    inv[e] = 1.0f / sL[wave * 16 + ((lane < 16) ? e : e + 8)];

  // Write context as f16 in [B,S,H,hd] == [B,S,D] for the output projection.
#pragma unroll
  for (int f = 0; f < 8; ++f)
#pragma unroll
    for (int e = 0; e < 8; ++e) {
      const int s = qrow0 + wave * 16 + ((lane < 16) ? e : e + 8);
      const int d = f * 16 + (lane & 15);
      Ctx[(((size_t)b * SEQ + s) * N_HEADS + h) * HEAD_DIM + d] =
          (_Float16)(o[f][e] * inv[e]);
    }
}

// ---------------------------------------------------------------------------
// Kernel 3: output projection (f16 A, f32 W->f16, f32 out + bias).
// ---------------------------------------------------------------------------
__global__ __launch_bounds__(256) void gemm_out(
    const _Float16* __restrict__ A, const float* __restrict__ W,
    const float* __restrict__ bias, float* __restrict__ out) {
  constexpr int BM = 128, BN = 128, BK = 32, LDK = BK + 8;
  __shared__ _Float16 sA[BM * LDK];
  __shared__ _Float16 sB[BN * LDK];

  const int m0 = blockIdx.y * BM;
  const int n0 = blockIdx.x * BN;
  const int tid = threadIdx.x;
  const int wave = tid >> 5, lane = tid & 31;
  const int wm = wave >> 1, wn = wave & 1;

  v8f acc[2][4];
#pragma unroll
  for (int i = 0; i < 2; ++i)
#pragma unroll
    for (int j = 0; j < 4; ++j) acc[i][j] = zero8();

  for (int kt = 0; kt < D_MODEL / BK; ++kt) {
    const int k0 = kt * BK;
    // Stage A (already f16): 128x32 halves as 512 8-half (b128) loads.
#pragma unroll
    for (int i = 0; i < 2; ++i) {
      const int idx = tid + i * 256;         // 0..511
      const int r = idx >> 2;                // 128 rows
      const int c = (idx & 3) * 8;           // 0,8,16,24
      *(v8h*)&sA[r * LDK + c] =
          *(const v8h*)&A[(size_t)(m0 + r) * D_MODEL + (k0 + c)];
    }
    // Stage B transposed (f32 -> f16): float4 loads, coalesced over n.
#pragma unroll
    for (int i = 0; i < 4; ++i) {
      const int idx = tid + i * 256;
      const int k = idx >> 5;
      const int n = (idx & 31) * 4;
      const v4f b4 = *(const v4f*)&W[(size_t)(k0 + k) * D_MODEL + (n0 + n)];
      sB[(n + 0) * LDK + k] = (_Float16)b4[0];
      sB[(n + 1) * LDK + k] = (_Float16)b4[1];
      sB[(n + 2) * LDK + k] = (_Float16)b4[2];
      sB[(n + 3) * LDK + k] = (_Float16)b4[3];
    }
    if (kt + 1 < D_MODEL / BK)
      __builtin_prefetch(&W[(size_t)(k0 + BK) * D_MODEL + n0 + (tid & 127)], 0, 3);
    __syncthreads();

    v16h af[2], bf[4];
#pragma unroll
    for (int mi = 0; mi < 2; ++mi)
      af[mi] = load_frag16(&sA[(wm * 32 + mi * 16) * LDK], LDK);
#pragma unroll
    for (int ni = 0; ni < 4; ++ni)
      bf[ni] = load_frag16(&sB[(wn * 64 + ni * 16) * LDK], LDK);
#pragma unroll
    for (int mi = 0; mi < 2; ++mi)
#pragma unroll
      for (int ni = 0; ni < 4; ++ni)
        acc[mi][ni] = wmma_f16(af[mi], bf[ni], acc[mi][ni]);
    __syncthreads();
  }

#pragma unroll
  for (int mi = 0; mi < 2; ++mi)
#pragma unroll
    for (int ni = 0; ni < 4; ++ni) {
      const int fr = m0 + wm * 32 + mi * 16;
      const int fc = n0 + wn * 64 + ni * 16;
#pragma unroll
      for (int e = 0; e < 8; ++e) {
        const int row = fr + ((lane < 16) ? e : e + 8);
        const int col = fc + (lane & 15);
        out[(size_t)row * D_MODEL + col] = acc[mi][ni][e] + bias[col];
      }
    }
}

// ---------------------------------------------------------------------------
extern "C" void kernel_launch(void* const* d_in, const int* in_sizes, int n_in,
                              void* d_out, int out_size, void* d_ws,
                              size_t ws_size, hipStream_t stream) {
  const float* x  = (const float*)d_in[0];
  const float* Wq = (const float*)d_in[1];
  const float* bq = (const float*)d_in[2];
  const float* Wk = (const float*)d_in[3];
  const float* bk = (const float*)d_in[4];
  const float* Wv = (const float*)d_in[5];
  const float* bv = (const float*)d_in[6];
  const float* Wo = (const float*)d_in[7];
  const float* bo = (const float*)d_in[8];
  float* out = (float*)d_out;

  const size_t elems = (size_t)M_ROWS * D_MODEL;  // 16,777,216 per buffer
  _Float16* Qh  = (_Float16*)d_ws;
  _Float16* Kh  = Qh + elems;
  _Float16* Vh  = Kh + elems;
  _Float16* Ctx = Vh + elems;   // total 128 MB of workspace

  // 1) QKV projections (z = 0:Q scaled, 1:K, 2:V)
  gemm_qkv<<<dim3(D_MODEL / 128, M_ROWS / 128, 3), 256, 0, stream>>>(
      x, Wq, Wk, Wv, bq, bk, bv, Qh, Kh, Vh);

  // 2) causal flash attention over (query-block, batch*head)
  flash_attn<<<dim3(SEQ / 64, BATCH * N_HEADS), 128, 0, stream>>>(Qh, Kh, Vh,
                                                                  Ctx);

  // 3) output projection
  gemm_out<<<dim3(D_MODEL / 128, M_ROWS / 128), 256, 0, stream>>>(Ctx, Wo, bo,
                                                                  out);
}
